// GatedAttention_34471407518326
// MI455X (gfx1250) — matrix operand order repaired and checked by reading, not compiled, not run on hardware
//
#include <hip/hip_runtime.h>
#include <hip/hip_bf16.h>

typedef __bf16 bf16_t;
typedef __attribute__((ext_vector_type(16))) __bf16 v16bf;
typedef __attribute__((ext_vector_type(8)))  float  v8f;
typedef int v4i32 __attribute__((vector_size(16)));

#define HIDN  1024
#define NH    8
#define NKV   2
#define HDIM  128
#define SEQ   2048
#define BATCH 4
#define NTOK  (BATCH*SEQ)

// Async global->LDS path (CDNA5 ASYNCcnt). Guarded so a missing builtin
// falls back to the register-bounce path instead of failing the compile.
#if defined(__gfx1250__) && \
    __has_builtin(__builtin_amdgcn_global_load_async_to_lds_b128) && \
    __has_builtin(__builtin_amdgcn_s_wait_asynccnt)
#define USE_ASYNC_LDS 1
#else
#define USE_ASYNC_LDS 0
#endif

#define AS1 __attribute__((address_space(1)))
#define AS3 __attribute__((address_space(3)))

#if USE_ASYNC_LDS
// Copy 16 bytes (8 bf16) global -> LDS via GLOBAL_LOAD_ASYNC_TO_LDS_B128.
// Builtin params are pointers to 16-byte int vectors (global, lds).
__device__ inline void async_cp16B(const bf16_t* g, bf16_t* l) {
  __builtin_amdgcn_global_load_async_to_lds_b128(
      (AS1 v4i32*)(g), (AS3 v4i32*)(l), 0, 0);
}
#endif

// ---------------------------------------------------------------------------
// WMMA helpers (CDNA5 gfx1250, wave32)
// ---------------------------------------------------------------------------
__device__ inline v8f vzero8() {
  v8f z = {0.f,0.f,0.f,0.f,0.f,0.f,0.f,0.f};
  return z;
}

__device__ inline v8f wmma_bf16(v16bf a, v16bf b, v8f c) {
  return __builtin_amdgcn_wmma_f32_16x16x32_bf16(false, a, false, b,
                                                 (short)0, c, false, false);
}

// A-fragment (16x32 bf16, MxK). Lane L covers row L&15;
// lanes<16 hold K 0..15; lanes>=16 hold K 16..31 (matches frag_b).
__device__ inline v16bf frag_a(const bf16_t* base, int stride, int lane) {
  int r  = lane & 15;
  int kb = (lane >> 4) << 4;
  const bf16_t* p = base + (size_t)r * stride + kb;
  union { uint4 v[2]; v16bf f; } u;
  u.v[0] = *(const uint4*)(p);
  u.v[1] = *(const uint4*)(p + 8);
  return u.f;
}

// B-fragment (32x16 bf16, KxN) from a TRANSPOSED tile (Bt[n][k], row stride
// `stride` elements). Lane L covers col n=L&15; lanes<16 hold K 0..15,
// lanes>=16 hold K 16..31 -> one contiguous 32B read.
__device__ inline v16bf frag_b(const bf16_t* baseT, int stride, int lane) {
  int n  = lane & 15;
  int kb = (lane >> 4) << 4;
  return *(const v16bf*)(baseT + (size_t)n * stride + kb);
}

__device__ inline float redmax16(float v) {
  v = fmaxf(v, __shfl_xor(v, 1, 32));
  v = fmaxf(v, __shfl_xor(v, 2, 32));
  v = fmaxf(v, __shfl_xor(v, 4, 32));
  v = fmaxf(v, __shfl_xor(v, 8, 32));
  return v;
}
__device__ inline float redsum16(float v) {
  v += __shfl_xor(v, 1, 32);
  v += __shfl_xor(v, 2, 32);
  v += __shfl_xor(v, 4, 32);
  v += __shfl_xor(v, 8, 32);
  return v;
}
__device__ inline float redsum32(float v) {
  v += __shfl_xor(v, 1, 32);
  v += __shfl_xor(v, 2, 32);
  v += __shfl_xor(v, 4, 32);
  v += __shfl_xor(v, 8, 32);
  v += __shfl_xor(v, 16, 32);
  return v;
}

// ---------------------------------------------------------------------------
// f32 -> bf16 convert
// ---------------------------------------------------------------------------
__global__ __launch_bounds__(256) void k_f2bf(const float* __restrict__ src,
                                              bf16_t* __restrict__ dst, int n) {
  int i = blockIdx.x * 256 + threadIdx.x;
  if (i < n) dst[i] = (bf16_t)src[i];
}

// ---------------------------------------------------------------------------
// Generic GEMM: C[M,N] = A[M,K](bf16) * B[K,N](bf16) + bias[N], f32 out.
// 128x128 tile per block (256 thr = 8 waves), BK=32; wave -> 32x64 sub-tile.
// M,N multiples of 128, K multiple of 32.
// ---------------------------------------------------------------------------
__global__ __launch_bounds__(256) void k_gemm_bf16(
    const bf16_t* __restrict__ A, const bf16_t* __restrict__ B,
    const float* __restrict__ bias, float* __restrict__ C,
    int M, int N, int K) {
  (void)M;
  __shared__ __align__(32) bf16_t sA[128 * 32];
  __shared__ __align__(32) bf16_t sBt[128 * 32];   // transposed: [n][k]

  int tid  = threadIdx.x;
  int lane = tid & 31;
  int wv   = tid >> 5;
  int m0   = blockIdx.x * 128;
  int n0   = blockIdx.y * 128;
  int wm   = (wv & 3) * 32;   // wave M offset within tile
  int wn   = (wv >> 2) * 64;  // wave N offset within tile

  v8f acc[2][4];
#pragma unroll
  for (int i = 0; i < 2; ++i)
#pragma unroll
    for (int j = 0; j < 4; ++j) acc[i][j] = vzero8();

  int arow = tid >> 1, aseg = (tid & 1) * 16;
  int bk   = tid >> 3, bseg = (tid & 7) * 16;

  for (int k0 = 0; k0 < K; k0 += 32) {
    // Stage A tile (128x32): contiguous 16B chunks -> async LDS copies.
    {
      const bf16_t* g = A + (size_t)(m0 + arow) * K + k0 + aseg;
      bf16_t* l = sA + arow * 32 + aseg;
#if USE_ASYNC_LDS
      async_cp16B(g, l);
      async_cp16B(g + 8, l + 8);
#else
      *(uint4*)(l)     = *(const uint4*)(g);
      *(uint4*)(l + 8) = *(const uint4*)(g + 8);
#endif
      if (k0 + 32 < K) __builtin_prefetch(g + 32, 0, 1);
    }
    // Stage B tile transposed into sBt[n][k] (scatter, cannot be async).
    {
      union { uint4 v[2]; bf16_t e[16]; } u;
      const bf16_t* g = B + (size_t)(k0 + bk) * N + n0 + bseg;
      u.v[0] = *(const uint4*)(g);
      u.v[1] = *(const uint4*)(g + 8);
#pragma unroll
      for (int j = 0; j < 16; ++j) sBt[(bseg + j) * 32 + bk] = u.e[j];
      if (k0 + 32 < K) __builtin_prefetch(g + (size_t)32 * N, 0, 1);
    }
#if USE_ASYNC_LDS
    __builtin_amdgcn_s_wait_asynccnt(0);
#endif
    __syncthreads();

    v16bf af[2], bfq[4];
#pragma unroll
    for (int i = 0; i < 2; ++i) af[i]  = frag_a(sA + (wm + i * 16) * 32, 32, lane);
#pragma unroll
    for (int j = 0; j < 4; ++j) bfq[j] = frag_b(sBt + (wn + j * 16) * 32, 32, lane);

#pragma unroll
    for (int i = 0; i < 2; ++i)
#pragma unroll
      for (int j = 0; j < 4; ++j)
        acc[i][j] = wmma_bf16(af[i], bfq[j], acc[i][j]);

    __syncthreads();
  }

  // C-frag layout: row = r + 8*(lane>>4), col = lane&15.
  int col = lane & 15;
  int ro  = (lane >> 4) * 8;
#pragma unroll
  for (int i = 0; i < 2; ++i)
#pragma unroll
    for (int j = 0; j < 4; ++j) {
      int cm = m0 + wm + i * 16 + ro;
      int cn = n0 + wn + j * 16 + col;
      float bb = bias[cn];
#pragma unroll
      for (int r = 0; r < 8; ++r)
        C[(size_t)(cm + r) * N + cn] = acc[i][j][r] + bb;
    }
}

// ---------------------------------------------------------------------------
// Q epilogue: rmsnorm over HD + rope, write bf16 (b,h,s,d).
// One wave per (token, head); 8 waves/block = 8 heads; grid = NTOK.
// ---------------------------------------------------------------------------
__global__ __launch_bounds__(256) void k_qnorm_rope(
    const float* __restrict__ qg, const float* __restrict__ rope,
    const float* __restrict__ w, bf16_t* __restrict__ qout) {
  int t = blockIdx.x;
  int h = threadIdx.x >> 5;
  int lane = threadIdx.x & 31;
  int s = t & (SEQ - 1);
  int b = t >> 11;
  int kv = h >> 2, g = h & 3;
  int d = lane * 4;

  const float* src = qg + (size_t)t * 2048 + kv * 1024 + g * 128 + d;
  float x0 = src[0], x1 = src[1], x2 = src[2], x3 = src[3];
  float ss = redsum32(x0 * x0 + x1 * x1 + x2 * x2 + x3 * x3);
  float inv = rsqrtf(ss * (1.f / 128.f) + 1e-6f);
  x0 = x0 * inv * w[d];     x1 = x1 * inv * w[d + 1];
  x2 = x2 * inv * w[d + 2]; x3 = x3 * inv * w[d + 3];

  const float* emb = rope + (size_t)s * 256;   // sin[0:128], cos[128:256]
  float o0 = x0 * emb[128 + d]     - x1 * emb[d];
  float o1 = x1 * emb[128 + d + 1] + x0 * emb[d + 1];
  float o2 = x2 * emb[128 + d + 2] - x3 * emb[d + 2];
  float o3 = x3 * emb[128 + d + 3] + x2 * emb[d + 3];

  bf16_t* dst = qout + (((size_t)b * NH + h) * SEQ + s) * HDIM + d;
  dst[0] = (bf16_t)o0; dst[1] = (bf16_t)o1;
  dst[2] = (bf16_t)o2; dst[3] = (bf16_t)o3;
}

// K epilogue: one wave per (token, kv). 8 waves/block = 4 tokens x 2 kv.
__global__ __launch_bounds__(256) void k_knorm_rope(
    const float* __restrict__ kf, const float* __restrict__ rope,
    const float* __restrict__ w, bf16_t* __restrict__ kout) {
  int lane = threadIdx.x & 31;
  int wv = threadIdx.x >> 5;
  int t = blockIdx.x * 4 + (wv >> 1);
  int kv = wv & 1;
  int s = t & (SEQ - 1);
  int b = t >> 11;
  int d = lane * 4;

  const float* src = kf + (size_t)t * 256 + kv * 128 + d;
  float x0 = src[0], x1 = src[1], x2 = src[2], x3 = src[3];
  float ss = redsum32(x0 * x0 + x1 * x1 + x2 * x2 + x3 * x3);
  float inv = rsqrtf(ss * (1.f / 128.f) + 1e-6f);
  x0 = x0 * inv * w[d];     x1 = x1 * inv * w[d + 1];
  x2 = x2 * inv * w[d + 2]; x3 = x3 * inv * w[d + 3];

  const float* emb = rope + (size_t)s * 256;
  float o0 = x0 * emb[128 + d]     - x1 * emb[d];
  float o1 = x1 * emb[128 + d + 1] + x0 * emb[d + 1];
  float o2 = x2 * emb[128 + d + 2] - x3 * emb[d + 2];
  float o3 = x3 * emb[128 + d + 3] + x2 * emb[d + 3];

  bf16_t* dst = kout + (((size_t)b * NKV + kv) * SEQ + s) * HDIM + d;
  dst[0] = (bf16_t)o0; dst[1] = (bf16_t)o1;
  dst[2] = (bf16_t)o2; dst[3] = (bf16_t)o3;
}

// V: (b,s,kv,d) f32 -> (b,kv,s,d) bf16
__global__ __launch_bounds__(256) void k_vpack(const float* __restrict__ vf,
                                               bf16_t* __restrict__ vout) {
  size_t idx = (size_t)blockIdx.x * 256 + threadIdx.x;  // < NTOK*256
  int d  = idx & 127;
  int kv = (idx >> 7) & 1;
  size_t t = idx >> 8;
  int s = (int)(t & (SEQ - 1));
  int b = (int)(t >> 11);
  vout[(((size_t)b * NKV + kv) * SEQ + s) * HDIM + d] = (bf16_t)vf[idx];
}

// ---------------------------------------------------------------------------
// Flash attention (non-causal). Block = (qtile=128, h, b), 8 waves; each
// wave owns 16 query rows. Keys processed in blocks of 32.
// ---------------------------------------------------------------------------
__global__ __launch_bounds__(256) void k_attn(
    const bf16_t* __restrict__ Q, const bf16_t* __restrict__ Kc,
    const bf16_t* __restrict__ Vc, float* __restrict__ O) {
  __shared__ __align__(32) bf16_t sK[32 * 128];    // [key][d]
  __shared__ __align__(32) bf16_t sVt[128 * 32];   // [d][key]
  __shared__ __align__(32) bf16_t sP[8 * 16 * 32]; // per-wave P tile

  int tid = threadIdx.x, lane = tid & 31, wv = tid >> 5;
  int qt = blockIdx.x, h = blockIdx.y, b = blockIdx.z;
  int kv = h >> 2;
  int m0 = qt * 128 + wv * 16;

  const bf16_t* qbase = Q + (((size_t)b * NH + h) * SEQ + m0) * HDIM;
  const bf16_t* kbase = Kc + ((size_t)b * NKV + kv) * SEQ * HDIM;
  const bf16_t* vbase = Vc + ((size_t)b * NKV + kv) * SEQ * HDIM;

  v16bf qf[4];
#pragma unroll
  for (int kc = 0; kc < 4; ++kc) qf[kc] = frag_a(qbase + kc * 32, HDIM, lane);

  v8f o[8];
  float m[8], l[8], alpha[8];
#pragma unroll
  for (int t2 = 0; t2 < 8; ++t2) o[t2] = vzero8();
#pragma unroll
  for (int r = 0; r < 8; ++r) { m[r] = -3.0e38f; l[r] = 0.f; }

  const float scale = 0.08838834764831845f;  // 1/sqrt(128)
  bf16_t* myP = sP + wv * 16 * 32;
  int krow = tid >> 3, seg = (tid & 7) * 16;

  for (int j = 0; j < SEQ; j += 32) {
    // Stage K block [32][128] (async path) and V block transposed [128][32].
    {
      const bf16_t* gk = kbase + ((size_t)j + krow) * HDIM + seg;
      bf16_t* lk = sK + krow * 128 + seg;
#if USE_ASYNC_LDS
      async_cp16B(gk, lk);
      async_cp16B(gk + 8, lk + 8);
#else
      *(uint4*)(lk)     = *(const uint4*)(gk);
      *(uint4*)(lk + 8) = *(const uint4*)(gk + 8);
#endif
      union { uint4 v[2]; bf16_t e[16]; } u;
      const bf16_t* gv = vbase + ((size_t)j + krow) * HDIM + seg;
      u.v[0] = *(const uint4*)(gv);
      u.v[1] = *(const uint4*)(gv + 8);
#pragma unroll
      for (int jj = 0; jj < 16; ++jj) sVt[(seg + jj) * 32 + krow] = u.e[jj];
      if (j + 32 < SEQ) {
        __builtin_prefetch(gk + 32 * HDIM, 0, 1);
        __builtin_prefetch(gv + 32 * HDIM, 0, 1);
      }
    }
#if USE_ASYNC_LDS
    __builtin_amdgcn_s_wait_asynccnt(0);
#endif
    __syncthreads();

    // S = Q * K^T : two 16x16 c-frags (key cols 0-15, 16-31).
    v8f s0 = vzero8(), s1 = vzero8();
#pragma unroll
    for (int kc = 0; kc < 4; ++kc) {
      v16bf b0 = frag_b(sK + kc * 32, 128, lane);
      v16bf b1 = frag_b(sK + 16 * 128 + kc * 32, 128, lane);
      s0 = wmma_bf16(qf[kc], b0, s0);
      s1 = wmma_bf16(qf[kc], b1, s1);
    }

    // Online softmax; C-frag row r maps to query row r + 8*(lane>>4).
#pragma unroll
    for (int r = 0; r < 8; ++r) {
      float s0r = s0[r] * scale;
      float s1r = s1[r] * scale;
      float mx = redmax16(fmaxf(s0r, s1r));
      float mn = fmaxf(m[r], mx);
      float al = __expf(m[r] - mn);
      float p0 = __expf(s0r - mn);
      float p1 = __expf(s1r - mn);
      float rs = redsum16(p0 + p1);
      l[r] = l[r] * al + rs;
      m[r] = mn;
      alpha[r] = al;
      int prow = r + ((lane >> 4) << 3);
      myP[prow * 32 + (lane & 15)]      = (bf16_t)p0;
      myP[prow * 32 + (lane & 15) + 16] = (bf16_t)p1;
    }

#pragma unroll
    for (int t2 = 0; t2 < 8; ++t2)
#pragma unroll
      for (int r = 0; r < 8; ++r) o[t2][r] *= alpha[r];

    // O += P * V : P as A-frag (16x32), V^T tiles as B-frags.
    v16bf pf = frag_a(myP, 32, lane);
#pragma unroll
    for (int t2 = 0; t2 < 8; ++t2) {
      v16bf bv = frag_b(sVt + t2 * 16 * 32, 32, lane);
      o[t2] = wmma_bf16(pf, bv, o[t2]);
    }
    __syncthreads();
  }

  int col = lane & 15, ro = (lane >> 4) * 8;
  float inv[8];
#pragma unroll
  for (int r = 0; r < 8; ++r) inv[r] = 1.f / l[r];
  float* obase = O + (((size_t)b * NH + h) * SEQ + m0) * HDIM;
#pragma unroll
  for (int t2 = 0; t2 < 8; ++t2)
#pragma unroll
    for (int r = 0; r < 8; ++r)
      obase[(size_t)(ro + r) * HDIM + t2 * 16 + col] = o[t2][r] * inv[r];
}

// ---------------------------------------------------------------------------
// Gate: x2[b,s,h*128+d] = bf16( ao[b,h,s,d] * sigmoid(gate) )
// ---------------------------------------------------------------------------
__global__ __launch_bounds__(256) void k_gatepack(
    const float* __restrict__ ao, const float* __restrict__ qg,
    bf16_t* __restrict__ x2) {
  size_t idx = (size_t)blockIdx.x * 256 + threadIdx.x;  // < NTOK*HIDN
  int d = idx & 127;
  int h = (int)((idx >> 7) & 7);
  size_t t = idx >> 10;
  int s = (int)(t & (SEQ - 1));
  int b = (int)(t >> 11);
  float a = ao[(((size_t)b * NH + h) * SEQ + s) * HDIM + d];
  float g = qg[t * 2048 + (h >> 2) * 1024 + 512 + (h & 3) * 128 + d];
  float sg = 1.f / (1.f + __expf(-g));
  x2[idx] = (bf16_t)(a * sg);
}

// ---------------------------------------------------------------------------
// Host launch
// ---------------------------------------------------------------------------
extern "C" void kernel_launch(void* const* d_in, const int* in_sizes, int n_in,
                              void* d_out, int out_size, void* d_ws, size_t ws_size,
                              hipStream_t stream) {
  (void)in_sizes; (void)n_in; (void)out_size; (void)ws_size;
  const float* hs  = (const float*)d_in[0];
  const float* rp  = (const float*)d_in[1];
  const float* Wq  = (const float*)d_in[2];
  const float* bq  = (const float*)d_in[3];
  const float* Wk  = (const float*)d_in[4];
  const float* bk  = (const float*)d_in[5];
  const float* Wv  = (const float*)d_in[6];
  const float* bv  = (const float*)d_in[7];
  const float* Wo  = (const float*)d_in[8];
  const float* bo  = (const float*)d_in[9];
  const float* qnw = (const float*)d_in[10];
  const float* knw = (const float*)d_in[11];
  float* out = (float*)d_out;

  char* base = (char*)d_ws;
  size_t off = 0;
  auto take = [&](size_t bytes) -> void* {
    void* p = base + off;
    off += (bytes + 255) & ~(size_t)255;
    return p;
  };
  bf16_t* hsb = (bf16_t*)take((size_t)NTOK * HIDN * 2);
  bf16_t* wqb = (bf16_t*)take((size_t)HIDN * 2048 * 2);
  bf16_t* wkb = (bf16_t*)take((size_t)HIDN * 256 * 2);
  bf16_t* wvb = (bf16_t*)take((size_t)HIDN * 256 * 2);
  bf16_t* wob = (bf16_t*)take((size_t)HIDN * HIDN * 2);
  float*  qg  = (float*) take((size_t)NTOK * 2048 * 4);
  float*  kf  = (float*) take((size_t)NTOK * 256 * 4);
  float*  vf  = (float*) take((size_t)NTOK * 256 * 4);
  bf16_t* qb  = (bf16_t*)take((size_t)NTOK * HIDN * 2);
  bf16_t* kb2 = (bf16_t*)take((size_t)NTOK * 256 * 2);
  bf16_t* vb2 = (bf16_t*)take((size_t)NTOK * 256 * 2);
  float*  ao  = (float*) take((size_t)NTOK * HIDN * 4);
  bf16_t* x2  = (bf16_t*)take((size_t)NTOK * HIDN * 2);

  dim3 blk(256);
  // 1) Convert to bf16
  k_f2bf<<<(NTOK * HIDN) / 256, blk, 0, stream>>>(hs, hsb, NTOK * HIDN);
  k_f2bf<<<(HIDN * 2048) / 256, blk, 0, stream>>>(Wq, wqb, HIDN * 2048);
  k_f2bf<<<(HIDN * 256) / 256, blk, 0, stream>>>(Wk, wkb, HIDN * 256);
  k_f2bf<<<(HIDN * 256) / 256, blk, 0, stream>>>(Wv, wvb, HIDN * 256);
  k_f2bf<<<(HIDN * HIDN) / 256, blk, 0, stream>>>(Wo, wob, HIDN * HIDN);

  // 2) Projections
  k_gemm_bf16<<<dim3(NTOK / 128, 2048 / 128), blk, 0, stream>>>(hsb, wqb, bq, qg, NTOK, 2048, HIDN);
  k_gemm_bf16<<<dim3(NTOK / 128, 256 / 128), blk, 0, stream>>>(hsb, wkb, bk, kf, NTOK, 256, HIDN);
  k_gemm_bf16<<<dim3(NTOK / 128, 256 / 128), blk, 0, stream>>>(hsb, wvb, bv, vf, NTOK, 256, HIDN);

  // 3) Norm + rope + layout
  k_qnorm_rope<<<NTOK, blk, 0, stream>>>(qg, rp, qnw, qb);
  k_knorm_rope<<<NTOK / 4, blk, 0, stream>>>(kf, rp, knw, kb2);
  k_vpack<<<(NTOK * 256) / 256, blk, 0, stream>>>(vf, vb2);

  // 4) Attention
  k_attn<<<dim3(SEQ / 128, NH, BATCH), blk, 0, stream>>>(qb, kb2, vb2, ao);

  // 5) Gate + output projection
  k_gatepack<<<(NTOK * HIDN) / 256, blk, 0, stream>>>(ao, qg, x2);
  k_gemm_bf16<<<dim3(NTOK / 128, HIDN / 128), blk, 0, stream>>>(x2, wob, bo, out, NTOK, HIDN, HIDN);
}
